// Attention_39015482917624
// MI455X (gfx1250) — matrix operand order repaired and checked by reading, not compiled, not run on hardware
//
#include <hip/hip_runtime.h>
#include <hip/hip_bf16.h>
#include <cstdint>

typedef __bf16 bf16;
typedef __attribute__((ext_vector_type(16))) __bf16 v16bf;
typedef __attribute__((ext_vector_type(8)))  __bf16 v8bf;
typedef __attribute__((ext_vector_type(8)))  float  v8f;
typedef __attribute__((ext_vector_type(4)))  float  v4f;

#define SQW   2048
#define SKW   2048
#define NB    8
#define DD    1024
#define HH    4
#define HDIM  32
#define INNER 128

static __device__ __forceinline__ v16bf cat8(v8bf a, v8bf b) {
  return __builtin_shufflevector(a, b, 0,1,2,3,4,5,6,7,8,9,10,11,12,13,14,15);
}
static __device__ __forceinline__ v8bf ldg8(const bf16* p) {
  return *reinterpret_cast<const v8bf*>(p);
}
static __device__ __forceinline__ v8f wmma_bf16(v16bf a, v16bf b, v8f c) {
  return __builtin_amdgcn_wmma_f32_16x16x32_bf16(false, a, false, b, (short)0, c,
                                                 false, false);
}

// Async global->LDS b128 (GVS mode: uniform 64-bit SGPR base + per-lane 32-bit
// byte offset; dst VGPR holds the LDS byte address = low 32 bits of the generic
// shared-space address per ISA 10.2). Tracked with ASYNCcnt.
static __device__ __forceinline__ void async_ld128(bf16* lds_dst, const bf16* gbase,
                                                   uint32_t byte_off) {
  uint32_t l = (uint32_t)(uintptr_t)lds_dst;
  asm volatile("global_load_async_to_lds_b128 %0, %1, %2"
               :: "v"(l), "v"(byte_off), "s"((uint64_t)(uintptr_t)gbase)
               : "memory");
}

// ---------------------------------------------------------------------------
// Kernel 1: convert all 4 weight matrices (each 131072 f32) to bf16 workspace
// ---------------------------------------------------------------------------
__global__ void cvt_weights_kernel(const float* __restrict__ Wq, const float* __restrict__ Wk,
                                   const float* __restrict__ Wv, const float* __restrict__ Wup,
                                   bf16* __restrict__ wq, bf16* __restrict__ wk,
                                   bf16* __restrict__ wv, bf16* __restrict__ wup) {
  int i = blockIdx.x * blockDim.x + threadIdx.x;   // exactly 131072 threads
  wq[i]  = (bf16)Wq[i];
  wk[i]  = (bf16)Wk[i];
  wv[i]  = (bf16)Wv[i];
  wup[i] = (bf16)Wup[i];
}

// ---------------------------------------------------------------------------
// Kernel 2: projection GEMM.  out = X(:,n,:) @ W^T  (bf16 WMMA, f32 acc)
//   X: (S, N, D) f32.  W: (128, 1024) bf16.
//   transpose_out==0:  out[((n*H+h)*S + s)*32 + c]   (Q, K)
//   transpose_out==1:  out[((n*H+h)*32 + c)*S + s]   (V, pre-transposed)
// ---------------------------------------------------------------------------
__global__ void __launch_bounds__(128)
proj_kernel(const float* __restrict__ X, const bf16* __restrict__ Wb,
            bf16* __restrict__ out, int S, int transpose_out) {
  const int lane = threadIdx.x & 31;
  const int wave = threadIdx.x >> 5;
  const int l  = lane & 15, hi = lane >> 4;
  const int groups = (S / 16) / 4;                 // 32
  const int n  = blockIdx.x / groups;
  const int g  = blockIdx.x % groups;
  const int s0 = (g * 4 + wave) * 16;

  const v8f vz = {0.f,0.f,0.f,0.f,0.f,0.f,0.f,0.f};
  v8f acc[8];
#pragma unroll
  for (int t = 0; t < 8; ++t) acc[t] = vz;

  const float* arow = X + ((size_t)(s0 + l) * NB + n) * DD;

  for (int k0 = 0; k0 < DD; k0 += 32) {
    const v4f* p0 = reinterpret_cast<const v4f*>(arow + k0 + hi * 8);
    const v4f* p1 = reinterpret_cast<const v4f*>(arow + k0 + 16 + hi * 8);
    v4f f0 = p0[0], f1 = p0[1], f2 = p1[0], f3 = p1[1];
    v16bf a;
#pragma unroll
    for (int i = 0; i < 4; ++i) {
      a[i]      = (bf16)f0[i];
      a[4 + i]  = (bf16)f1[i];
      a[8 + i]  = (bf16)f2[i];
      a[12 + i] = (bf16)f3[i];
    }
#pragma unroll
    for (int t = 0; t < 8; ++t) {
      const bf16* brow = Wb + (size_t)(t * 16 + l) * DD + k0 + hi * 8;
      v16bf b = cat8(ldg8(brow), ldg8(brow + 16));
      acc[t] = wmma_bf16(a, b, acc[t]);
    }
  }

#pragma unroll
  for (int t = 0; t < 8; ++t) {
    const int e = t * 16 + l;
    const int h = e >> 5, c = e & 31;
#pragma unroll
    for (int r = 0; r < 8; ++r) {
      const int s = s0 + hi * 8 + r;
      size_t idx = transpose_out
          ? (((size_t)(n * HH + h) * HDIM + c) * (size_t)S + s)
          : (((size_t)(n * HH + h) * (size_t)S + s) * HDIM + c);
      out[idx] = (bf16)acc[t][r];
    }
  }
}

// ---------------------------------------------------------------------------
// Kernel 3: flash attention, 4 waves/block sharing one (n,h).
//   K/V 32-key chunks staged to LDS with async-to-LDS, double buffered.
//   Qb,Kb: (N*H, S, 32) bf16 ; Vt: (N*H, 32, Sk) bf16 ; Xb: (N, Sq, 128) bf16
// ---------------------------------------------------------------------------
__global__ void __launch_bounds__(128)
attn_kernel(const bf16* __restrict__ Qb, const bf16* __restrict__ Kb,
            const bf16* __restrict__ Vt, bf16* __restrict__ Xb) {
  __shared__ alignas(16) bf16 kbuf[2][32 * 32];   // [key_local][c]
  __shared__ alignas(16) bf16 vbuf[2][32 * 32];   // [c][key_local]
  __shared__ alignas(16) bf16 pbuf[4][16 * 32];   // per-wave P transpose

  const int tid  = threadIdx.x;
  const int lane = tid & 31;
  const int wave = tid >> 5;
  const int l = lane & 15, hi = lane >> 4;
  const int nh = blockIdx.x >> 5;                 // 0..31
  const int g  = blockIdx.x & 31;                 // 0..31
  const int n = nh >> 2, h = nh & 3;
  const int s0 = (g * 4 + wave) * 16;
  const float scale = 0.17677669529663687f;       // 32^-0.5

  const bf16* kbase = Kb + (size_t)nh * SKW * HDIM;
  const bf16* vbase = Vt + (size_t)nh * HDIM * SKW;

  // staging assignment: thread -> (row 0..31, 16B chunk 0..3)
  const int srow = tid >> 2, sch = (tid & 3) * 8;

  // Q tile, A layout (resident for the whole loop)
  const bf16* qrow = Qb + ((size_t)nh * SQW + (s0 + l)) * HDIM;
  v16bf qa = cat8(ldg8(qrow + hi * 8), ldg8(qrow + 16 + hi * 8));

  const v8f vz = {0.f,0.f,0.f,0.f,0.f,0.f,0.f,0.f};
  v8f o0 = vz, o1 = vz;
  float rm[8], rl[8];
#pragma unroll
  for (int r = 0; r < 8; ++r) { rm[r] = -3.0e38f; rl[r] = 0.0f; }

  // prologue: async-stage chunk 0 into buffer 0 (2 async instrs per wave)
  async_ld128(&kbuf[0][srow * 32 + sch], kbase,
              (uint32_t)(((0 + srow) * HDIM + sch) * sizeof(bf16)));
  async_ld128(&vbuf[0][srow * 32 + sch], vbase,
              (uint32_t)((srow * SKW + 0 + sch) * sizeof(bf16)));

  const int NC = SKW / 32;                        // 64 chunks
  for (int ic = 0; ic < NC; ++ic) {
    const int buf = ic & 1;
    if (ic + 1 < NC) {
      const int kc1 = (ic + 1) * 32;
      async_ld128(&kbuf[buf ^ 1][srow * 32 + sch], kbase,
                  (uint32_t)(((kc1 + srow) * HDIM + sch) * sizeof(bf16)));
      async_ld128(&vbuf[buf ^ 1][srow * 32 + sch], vbase,
                  (uint32_t)((srow * SKW + kc1 + sch) * sizeof(bf16)));
      asm volatile("s_wait_asynccnt 0x2" ::: "memory");   // chunk ic landed
    } else {
      asm volatile("s_wait_asynccnt 0x0" ::: "memory");
    }
    __syncthreads();                              // chunk ic visible to all waves

    // S = Q @ K^T  (B from LDS: column j = key row, c-elems in table order)
    const bf16* kr0 = &kbuf[buf][l * 32 + hi * 8];
    const bf16* kr1 = &kbuf[buf][(16 + l) * 32 + hi * 8];
    v16bf kb0 = cat8(*(const v8bf*)kr0, *(const v8bf*)(kr0 + 16));
    v16bf kb1 = cat8(*(const v8bf*)kr1, *(const v8bf*)(kr1 + 16));
    v8f st0 = wmma_bf16(qa, kb0, vz);
    v8f st1 = wmma_bf16(qa, kb1, vz);

    float p0v[8], p1v[8];
#pragma unroll
    for (int r = 0; r < 8; ++r) {
      float x0 = st0[r] * scale, x1 = st1[r] * scale;
      float mx = fmaxf(x0, x1);
      mx = fmaxf(mx, __shfl_xor(mx, 1));
      mx = fmaxf(mx, __shfl_xor(mx, 2));
      mx = fmaxf(mx, __shfl_xor(mx, 4));
      mx = fmaxf(mx, __shfl_xor(mx, 8));
      float mnew = fmaxf(rm[r], mx);
      float p0 = __expf(x0 - mnew);
      float p1 = __expf(x1 - mnew);
      float rs = p0 + p1;
      rs += __shfl_xor(rs, 1);
      rs += __shfl_xor(rs, 2);
      rs += __shfl_xor(rs, 4);
      rs += __shfl_xor(rs, 8);
      float al = __expf(rm[r] - mnew);
      rl[r] = rl[r] * al + rs;
      rm[r] = mnew;
      o0[r] *= al;
      o1[r] *= al;
      p0v[r] = p0; p1v[r] = p1;
    }

    // P: C/D layout -> A layout via per-wave LDS tile
    bf16* pb = pbuf[wave];
#pragma unroll
    for (int r = 0; r < 8; ++r) {
      pb[(hi * 8 + r) * 32 + l]      = (bf16)p0v[r];
      pb[(hi * 8 + r) * 32 + 16 + l] = (bf16)p1v[r];
    }
    __syncthreads();
    const bf16* prow = pb + l * 32;
    v16bf pa = cat8(*(const v8bf*)(prow + hi * 8),
                    *(const v8bf*)(prow + 16 + hi * 8));

    // O += P @ V  (B from LDS: column c, key-elems in table order)
    const bf16* vr0 = &vbuf[buf][l * 32 + hi * 8];
    const bf16* vr1 = &vbuf[buf][(16 + l) * 32 + hi * 8];
    v16bf vb0 = cat8(*(const v8bf*)vr0, *(const v8bf*)(vr0 + 16));
    v16bf vb1 = cat8(*(const v8bf*)vr1, *(const v8bf*)(vr1 + 16));
    o0 = wmma_bf16(pa, vb0, o0);
    o1 = wmma_bf16(pa, vb1, o1);

    __syncthreads();   // all waves done reading buf before it is re-filled
  }

  // normalize rows and store x: (N, Sq, INNER), e = h*32 + c
#pragma unroll
  for (int r = 0; r < 8; ++r) {
    const float inv = 1.0f / rl[r];
    const int s = s0 + hi * 8 + r;
    bf16* xrow = Xb + ((size_t)n * SQW + s) * INNER + h * HDIM;
    xrow[l]      = (bf16)(o0[r] * inv);
    xrow[16 + l] = (bf16)(o1[r] * inv);
  }
}

// ---------------------------------------------------------------------------
// Kernel 4: up-projection. out[(s*N+n)*D + d] = sum_e x[n,s,e] * Wup[d,e]
// ---------------------------------------------------------------------------
__global__ void __launch_bounds__(128)
upproj_kernel(const bf16* __restrict__ Xb, const bf16* __restrict__ Wupb,
              float* __restrict__ out) {
  const int lane = threadIdx.x & 31;
  const int wave = threadIdx.x >> 5;
  const int l = lane & 15, hi = lane >> 4;
  const int bq  = blockIdx.x;
  const int n   = bq >> 9;
  const int rem = bq & 511;
  const int st  = rem >> 2;
  const int dq  = rem & 3;
  const int d0  = (dq * 4 + wave) * 64;
  const int s0  = st * 16;

  const v8f vz = {0.f,0.f,0.f,0.f,0.f,0.f,0.f,0.f};
  v8f acc[4];
#pragma unroll
  for (int t = 0; t < 4; ++t) acc[t] = vz;

  const bf16* arow = Xb + ((size_t)n * SQW + (s0 + l)) * INNER;

#pragma unroll
  for (int e0 = 0; e0 < INNER; e0 += 32) {
    v16bf a = cat8(ldg8(arow + e0 + hi * 8), ldg8(arow + e0 + 16 + hi * 8));
#pragma unroll
    for (int t = 0; t < 4; ++t) {
      const bf16* brow = Wupb + (size_t)(d0 + t * 16 + l) * INNER + e0 + hi * 8;
      v16bf b = cat8(ldg8(brow), ldg8(brow + 16));
      acc[t] = wmma_bf16(a, b, acc[t]);
    }
  }

#pragma unroll
  for (int t = 0; t < 4; ++t) {
    const int d = d0 + t * 16 + l;
#pragma unroll
    for (int r = 0; r < 8; ++r) {
      const int s = s0 + hi * 8 + r;
      out[((size_t)s * NB + n) * DD + d] = acc[t][r];
    }
  }
}

// ---------------------------------------------------------------------------
extern "C" void kernel_launch(void* const* d_in, const int* in_sizes, int n_in,
                              void* d_out, int out_size, void* d_ws, size_t ws_size,
                              hipStream_t stream) {
  const float* query = (const float*)d_in[0];
  const float* keyf  = (const float*)d_in[1];
  const float* Wq    = (const float*)d_in[2];
  const float* Wk    = (const float*)d_in[3];
  const float* Wv    = (const float*)d_in[4];
  const float* Wup   = (const float*)d_in[5];
  float* out = (float*)d_out;

  bf16* ws = (bf16*)d_ws;
  const size_t WSZ = (size_t)INNER * DD;                 // 131072 per weight
  const size_t PSZ = (size_t)NB * HH * SQW * HDIM;       // 2097152 per tensor
  bf16* wqb  = ws;
  bf16* wkb  = wqb + WSZ;
  bf16* wvb  = wkb + WSZ;
  bf16* wupb = wvb + WSZ;
  bf16* Qb   = wupb + WSZ;
  bf16* Kb   = Qb + PSZ;
  bf16* Vt   = Kb + PSZ;
  bf16* Xb   = Vt + PSZ;

  cvt_weights_kernel<<<512, 256, 0, stream>>>(Wq, Wk, Wv, Wup, wqb, wkb, wvb, wupb);
  proj_kernel<<<256, 128, 0, stream>>>(query, wqb, Qb, SQW, 0);
  proj_kernel<<<256, 128, 0, stream>>>(keyf,  wkb, Kb, SKW, 0);
  proj_kernel<<<256, 128, 0, stream>>>(keyf,  wvb, Vt, SKW, 1);
  attn_kernel<<<32 * 32, 128, 0, stream>>>(Qb, Kb, Vt, Xb);
  upproj_kernel<<<NB * (SQW / 16) * 4, 128, 0, stream>>>(Xb, wupb, out);
}
